// WindowTransformerBlock_22617297781116
// MI455X (gfx1250) — compile-verified, matrix operand order checked by reading, not run on hardware
//
#include <hip/hip_runtime.h>
#include <hip/hip_bf16.h>
#include <stdint.h>

// ---------------- problem constants ----------------
#define Bk    8
#define Ck    192
#define Hk    128
#define Wk    128
#define WSk   8
#define SSk   4
#define NHEAD 6
#define HDIM  32
#define HIDk  768
#define NTOK  64                 // tokens per window (8x8)
#define NWIN  2048               // total windows = 8 * 16 * 16
#define MROWS (NWIN * NTOK)      // 131072 token rows
static constexpr float SCALEQ = 0.17677669529663687f;  // 32^-0.5
#define EPSk 1e-5f

// ---------------- WMMA types ----------------
typedef __attribute__((ext_vector_type(16))) __bf16 v16bf;
typedef __attribute__((ext_vector_type(8)))  float  v8f;

union FragU { v16bf v; uint4 q[2]; };

__device__ inline unsigned short f2bf(float x) {
  union { float f; unsigned int u; } c; c.f = x;
  unsigned int u = c.u;
  unsigned int r = (u + 0x7FFFu + ((u >> 16) & 1u)) >> 16;
  return (unsigned short)r;
}

// Fragment layout for v_wmma_f32_16x16x32_bf16 (ISA 7.12.2):
// row/col = lane&15; element e<8 -> K = (lane>>4)*8 + e,
// element e>=8 -> K = 16 + (lane>>4)*8 + (e-8). Two b128 loads per lane.
// Uniform base + 32-bit element offset -> SADDR+voffset global loads
// (1 offset VGPR per stream instead of a 64-bit pointer pair).
__device__ inline v16bf frag_off(const unsigned short* __restrict__ base,
                                 unsigned off) {
  FragU f;
  f.q[0] = *(const uint4*)(base + off);
  f.q[1] = *(const uint4*)(base + off + 16);
  return f.v;
}

__device__ inline v8f wmma_bf16(v16bf a, v16bf b, v8f c) {
  return __builtin_amdgcn_wmma_f32_16x16x32_bf16(false, a, false, b,
                                                 (short)0, c, false, false);
}

__device__ inline float gelu_exact(float x) {
  return 0.5f * x * (1.0f + erff(x * 0.70710678118654752f));
}

// ---------------- kernel 0: weight transpose + bf16 convert ----------------
// Win: [K][N] f32 row-major  ->  Wt: [N][K] bf16 row-major
__global__ void wt_transpose_kernel(const float* __restrict__ Win,
                                    unsigned short* __restrict__ Wt,
                                    int K, int N) {
  int idx = blockIdx.x * blockDim.x + threadIdx.x;
  if (idx >= K * N) return;
  int n = idx / K;
  int k = idx - n * K;
  Wt[idx] = f2bf(Win[(size_t)k * N + n]);
}

// ---------------- kernel 1: dwconv residual + roll + window partition ----------
// out windows f32: [win][tok][c], win = b*256 + wh*16 + ww
__global__ void dwconv_partition_kernel(const float* __restrict__ x,
                                        const float* __restrict__ cw,
                                        float* __restrict__ windows) {
  int idx = blockIdx.x * blockDim.x + threadIdx.x;
  if (idx >= NWIN * NTOK * Ck) return;
  int w = idx / (NTOK * Ck);
  int rem = idx - w * (NTOK * Ck);
  int t = rem / Ck;
  int c = rem - t * Ck;
  int b  = w >> 8;
  int wi = w & 255;
  int wh = wi >> 4, ww = wi & 15;
  int hr = wh * WSk + (t >> 3);          // rolled-frame coords
  int wr = ww * WSk + (t & 7);
  int hs = (hr + SSk) & (Hk - 1);        // pre-roll coords
  int ws = (wr + SSk) & (Wk - 1);
  const float* xb = x + ((size_t)(b * Ck + c) * Hk) * Wk;
  float acc = xb[hs * Wk + ws];          // residual
  #pragma unroll
  for (int kh = 0; kh < 3; ++kh) {
    int ih = hs + kh - 1;
    if (ih < 0 || ih >= Hk) continue;
    #pragma unroll
    for (int kw = 0; kw < 3; ++kw) {
      int iw = ws + kw - 1;
      if (iw < 0 || iw >= Wk) continue;
      acc += cw[c * 9 + kh * 3 + kw] * xb[ih * Wk + iw];
    }
  }
  windows[idx] = acc;
}

// ---------------- kernel 2: LayerNorm over C=192, wave per token -> bf16 -----
__global__ void layernorm_kernel(const float* __restrict__ in,
                                 const float* __restrict__ g,
                                 const float* __restrict__ bta,
                                 unsigned short* __restrict__ out) {
  int token = blockIdx.x * (blockDim.x >> 5) + (threadIdx.x >> 5);
  int lane  = threadIdx.x & 31;
  if (token >= MROWS) return;
  const float* row = in + (size_t)token * Ck;
  float v[6];
  float s = 0.f;
  #pragma unroll
  for (int j = 0; j < 6; ++j) { v[j] = row[lane + 32 * j]; s += v[j]; }
  #pragma unroll
  for (int o = 16; o > 0; o >>= 1) s += __shfl_xor(s, o, 32);
  float mu = s * (1.0f / Ck);
  float vs = 0.f;
  #pragma unroll
  for (int j = 0; j < 6; ++j) { float d = v[j] - mu; vs += d * d; }
  #pragma unroll
  for (int o = 16; o > 0; o >>= 1) vs += __shfl_xor(vs, o, 32);
  float rstd = rsqrtf(vs * (1.0f / Ck) + EPSk);
  unsigned short* orow = out + (size_t)token * Ck;
  #pragma unroll
  for (int j = 0; j < 6; ++j) {
    int c = lane + 32 * j;
    orow[c] = f2bf((v[j] - mu) * rstd * g[c] + bta[c]);
  }
}

// ---------------- kernel 3: generic bf16 WMMA GEMM, 32x64 tile per wave ------
// Software-pipelined K loop (double-buffered fragments) with uniform-base +
// 32-bit-offset addressing to keep VGPR pressure low (no spills).
// mode 0: out_f32 = acc + bias (+ residual)
// mode 1: QKV scatter: q (scaled, bf16), k (bf16), v transposed (bf16)
// mode 2: out_bf16 = gelu(acc + bias)
#define GM_F32  0
#define GM_QKV  1
#define GM_GELU 2

__global__ void gemm_bf16_kernel(const unsigned short* __restrict__ A,
                                 const unsigned short* __restrict__ Bt,
                                 const float* __restrict__ bias,
                                 int M, int N, int K, int mode,
                                 float* __restrict__ outf,
                                 unsigned short* __restrict__ outb,
                                 const float* __restrict__ residual,
                                 unsigned short* __restrict__ qo,
                                 unsigned short* __restrict__ ko,
                                 unsigned short* __restrict__ vto) {
  int waveG = blockIdx.x * (blockDim.x >> 5) + (threadIdx.x >> 5);
  int lane  = threadIdx.x & 31;
  int tilesN = N >> 6;
  int mT = waveG / tilesN;
  int nT = waveG - mT * tilesN;
  int tileM = mT * 32;
  int tileN = nT * 64;
  if (tileM >= M) return;

  // 32-bit element offsets (base pointers stay uniform in SGPRs)
  unsigned swz  = (unsigned)((lane >> 4) << 3);
  unsigned rowK = (unsigned)(lane & 15) * (unsigned)K;
  unsigned oa0 = (unsigned)tileM * (unsigned)K + rowK + swz;
  unsigned oa1 = oa0 + 16u * (unsigned)K;
  unsigned ob0 = (unsigned)tileN * (unsigned)K + rowK + swz;
  unsigned ob1 = ob0 + 16u * (unsigned)K;
  unsigned ob2 = ob1 + 16u * (unsigned)K;
  unsigned ob3 = ob2 + 16u * (unsigned)K;

  v8f acc[2][4];
  #pragma unroll
  for (int i = 0; i < 2; ++i)
    #pragma unroll
    for (int j = 0; j < 4; ++j) acc[i][j] = (v8f){0.f,0.f,0.f,0.f,0.f,0.f,0.f,0.f};

  // prologue: fragments for k0 = 0
  v16bf a0 = frag_off(A, oa0), a1 = frag_off(A, oa1);
  v16bf b0 = frag_off(Bt, ob0), b1 = frag_off(Bt, ob1);
  v16bf b2 = frag_off(Bt, ob2), b3 = frag_off(Bt, ob3);

  for (unsigned k0 = 32; k0 < (unsigned)K; k0 += 32) {
    // issue next-step loads first (stay in flight across the WMMAs below)
    v16bf na0 = frag_off(A, oa0 + k0), na1 = frag_off(A, oa1 + k0);
    v16bf nb0 = frag_off(Bt, ob0 + k0), nb1 = frag_off(Bt, ob1 + k0);
    v16bf nb2 = frag_off(Bt, ob2 + k0), nb3 = frag_off(Bt, ob3 + k0);

    acc[0][0] = wmma_bf16(a0, b0, acc[0][0]);
    acc[1][0] = wmma_bf16(a1, b0, acc[1][0]);
    acc[0][1] = wmma_bf16(a0, b1, acc[0][1]);
    acc[1][1] = wmma_bf16(a1, b1, acc[1][1]);
    acc[0][2] = wmma_bf16(a0, b2, acc[0][2]);
    acc[1][2] = wmma_bf16(a1, b2, acc[1][2]);
    acc[0][3] = wmma_bf16(a0, b3, acc[0][3]);
    acc[1][3] = wmma_bf16(a1, b3, acc[1][3]);

    a0 = na0; a1 = na1; b0 = nb0; b1 = nb1; b2 = nb2; b3 = nb3;
  }
  // epilogue K-step
  acc[0][0] = wmma_bf16(a0, b0, acc[0][0]);
  acc[1][0] = wmma_bf16(a1, b0, acc[1][0]);
  acc[0][1] = wmma_bf16(a0, b1, acc[0][1]);
  acc[1][1] = wmma_bf16(a1, b1, acc[1][1]);
  acc[0][2] = wmma_bf16(a0, b2, acc[0][2]);
  acc[1][2] = wmma_bf16(a1, b2, acc[1][2]);
  acc[0][3] = wmma_bf16(a0, b3, acc[0][3]);
  acc[1][3] = wmma_bf16(a1, b3, acc[1][3]);

  #pragma unroll
  for (int mi = 0; mi < 2; ++mi) {
    #pragma unroll
    for (int ni = 0; ni < 4; ++ni) {
      int col = tileN + ni * 16 + (lane & 15);
      float bv = bias ? bias[col] : 0.f;
      // QKV column decode (uniform per (ni,lane))
      int which = col / Ck;
      int hc    = col - which * Ck;
      int head  = hc >> 5;
      int d     = hc & 31;
      #pragma unroll
      for (int i = 0; i < 8; ++i) {
        int row = tileM + mi * 16 + ((lane >> 4) << 3) + i;
        float val = acc[mi][ni][i] + bv;
        if (mode == GM_F32) {
          if (residual) val += residual[(size_t)row * N + col];
          outf[(size_t)row * N + col] = val;
        } else if (mode == GM_GELU) {
          outb[(size_t)row * N + col] = f2bf(gelu_exact(val));
        } else { // GM_QKV
          int w = row >> 6;
          int t = row & 63;
          size_t wh = (size_t)(w * NHEAD + head);
          if (which == 0)
            qo[(wh * NTOK + t) * HDIM + d] = f2bf(val * SCALEQ);
          else if (which == 1)
            ko[(wh * NTOK + t) * HDIM + d] = f2bf(val);
          else
            vto[(wh * HDIM + d) * NTOK + t] = f2bf(val);   // V transposed [d][t]
        }
      }
    }
  }
}

// ---------------- kernel 4: windowed attention, 1 WG (4 waves) per (win,head) -
__global__ void attention_kernel(const unsigned short* __restrict__ q,
                                 const unsigned short* __restrict__ k,
                                 const unsigned short* __restrict__ vt,
                                 const float* __restrict__ bias_table,
                                 unsigned short* __restrict__ attn_out) {
  __shared__ __align__(16) float S[64 * 65];          // padded: no bank conflicts
  __shared__ __align__(16) unsigned short P[64 * 72]; // bf16 probs, 16B-aligned rows

  int wid  = blockIdx.x;              // (window, head) pair
  int w    = wid / NHEAD;
  int head = wid - w * NHEAD;
  int lane = threadIdx.x & 31;
  int wave = threadIdx.x >> 5;
  int m0   = wave * 16;

  unsigned swz = (unsigned)((lane >> 4) << 3);
  unsigned qkbase = (unsigned)wid * (NTOK * HDIM);

  int wi = w & 255;
  int wh = wi >> 4, ww = wi & 15;

  // ---- S = (q*scale) . k^T  (64x64, K=32) : 4 WMMAs per wave ----
  v16bf aq = frag_off(q, qkbase + (unsigned)(m0 + (lane & 15)) * HDIM + swz);
  unsigned okb = qkbase + (unsigned)(lane & 15) * HDIM + swz;
  #pragma unroll
  for (int nt = 0; nt < 4; ++nt) {
    v16bf bk = frag_off(k, okb + (unsigned)(nt * 16 * HDIM));
    v8f accS = (v8f){0.f,0.f,0.f,0.f,0.f,0.f,0.f,0.f};
    accS = wmma_bf16(aq, bk, accS);
    int col = nt * 16 + (lane & 15);          // key token j
    int jh = col >> 3, jw = col & 7;
    int gjh = wh * WSk + jh, gjw = ww * WSk + jw;
    int jreg = (gjh < 120 ? 0 : (gjh < 124 ? 1 : 2)) * 3 +
               (gjw < 120 ? 0 : (gjw < 124 ? 1 : 2));
    #pragma unroll
    for (int i = 0; i < 8; ++i) {
      int rtok = m0 + ((lane >> 4) << 3) + i; // query token i
      int ih = rtok >> 3, iw = rtok & 7;
      float bias = bias_table[((ih - jh + 7) * 15 + (iw - jw + 7)) * NHEAD + head];
      int gih = wh * WSk + ih, giw = ww * WSk + iw;
      int ireg = (gih < 120 ? 0 : (gih < 124 ? 1 : 2)) * 3 +
                 (giw < 120 ? 0 : (giw < 124 ? 1 : 2));
      float msk = (ireg == jreg) ? 0.f : -100.f;
      S[rtok * 65 + col] = accS[i] + bias + msk;
    }
  }
  __syncthreads();

  // ---- row softmax -> bf16 probs in LDS ----
  if (threadIdx.x < 64) {
    int r = threadIdx.x;
    float mx = -1e30f;
    for (int j = 0; j < 64; ++j) mx = fmaxf(mx, S[r * 65 + j]);
    float sum = 0.f;
    for (int j = 0; j < 64; ++j) {
      float e = expf(S[r * 65 + j] - mx);
      S[r * 65 + j] = e;
      sum += e;
    }
    float inv = 1.0f / sum;
    for (int j = 0; j < 64; ++j) P[r * 72 + j] = f2bf(S[r * 65 + j] * inv);
  }
  __syncthreads();

  // ---- O = P @ V (64x32, K=64): 2 N-tiles x 2 K-steps per wave ----
  unsigned opp = (unsigned)(m0 + (lane & 15)) * 72 + swz;
  unsigned ovb = qkbase + (unsigned)(lane & 15) * NTOK + swz;
  #pragma unroll
  for (int nt = 0; nt < 2; ++nt) {
    v8f accO = (v8f){0.f,0.f,0.f,0.f,0.f,0.f,0.f,0.f};
    #pragma unroll
    for (unsigned k0 = 0; k0 < 64; k0 += 32) {
      v16bf pa = frag_off((const unsigned short*)P, opp + k0);
      v16bf bv = frag_off(vt, ovb + (unsigned)(nt * 16 * NTOK) + k0);
      accO = wmma_bf16(pa, bv, accO);
    }
    int d = nt * 16 + (lane & 15);
    #pragma unroll
    for (int i = 0; i < 8; ++i) {
      int rtok = m0 + ((lane >> 4) << 3) + i;
      attn_out[((size_t)(w * NTOK + rtok)) * Ck + head * HDIM + d] = f2bf(accO[i]);
    }
  }
}

// ---------------- kernel 5: unpartition + crop + roll back -------------------
__global__ void unpartition_kernel(const float* __restrict__ win3,
                                   float* __restrict__ out) {
  int idx = blockIdx.x * blockDim.x + threadIdx.x;
  if (idx >= Bk * Ck * Hk * Wk) return;
  int b = idx / (Ck * Hk * Wk);
  int rem = idx - b * (Ck * Hk * Wk);
  int c = rem / (Hk * Wk);
  rem -= c * (Hk * Wk);
  int h = rem / Wk;
  int x = rem - h * Wk;
  int hr = (h - SSk + Hk) & (Hk - 1);
  int wr = (x - SSk + Wk) & (Wk - 1);
  int w = b * 256 + (hr >> 3) * 16 + (wr >> 3);
  int t = (hr & 7) * WSk + (wr & 7);
  out[idx] = win3[((size_t)w * NTOK + t) * Ck + c];
}

// ---------------- host orchestration ----------------
extern "C" void kernel_launch(void* const* d_in, const int* in_sizes, int n_in,
                              void* d_out, int out_size, void* d_ws, size_t ws_size,
                              hipStream_t stream) {
  const float* x        = (const float*)d_in[0];
  const float* convw    = (const float*)d_in[1];
  const float* n1g      = (const float*)d_in[2];
  const float* n1b      = (const float*)d_in[3];
  const float* qkv_w    = (const float*)d_in[4];
  const float* qkv_b    = (const float*)d_in[5];
  const float* proj_w   = (const float*)d_in[6];
  const float* proj_b   = (const float*)d_in[7];
  const float* relbias  = (const float*)d_in[8];
  const float* n2g      = (const float*)d_in[9];
  const float* n2b      = (const float*)d_in[10];
  const float* mlp_w1   = (const float*)d_in[11];
  const float* mlp_b1   = (const float*)d_in[12];
  const float* mlp_w2   = (const float*)d_in[13];
  const float* mlp_b2   = (const float*)d_in[14];
  float* out = (float*)d_out;

  // workspace carve-out
  char* p = (char*)d_ws;
  auto carve = [&](size_t bytes) -> void* {
    void* r = (void*)p;
    p += (bytes + 255) & ~((size_t)255);
    return r;
  };
  const size_t TOK = (size_t)MROWS;
  float*          windows  = (float*)         carve(TOK * Ck * 4);  // f32 residual
  unsigned short* ln1      = (unsigned short*)carve(TOK * Ck * 2);
  unsigned short* qbuf     = (unsigned short*)carve(TOK * Ck * 2);  // [win*head][t][d]
  unsigned short* kbuf     = (unsigned short*)carve(TOK * Ck * 2);
  unsigned short* vtbuf    = (unsigned short*)carve(TOK * Ck * 2);  // transposed
  unsigned short* attn_out = (unsigned short*)carve(TOK * Ck * 2);
  float*          windows2 = (float*)         carve(TOK * Ck * 4);
  unsigned short* ln2      = (unsigned short*)carve(TOK * Ck * 2);
  unsigned short* hbuf     = (unsigned short*)carve(TOK * HIDk * 2);
  float*          windows3 = (float*)         carve(TOK * Ck * 4);
  unsigned short* qkv_wt   = (unsigned short*)carve((size_t)Ck * 3 * Ck * 2);
  unsigned short* proj_wt  = (unsigned short*)carve((size_t)Ck * Ck * 2);
  unsigned short* mlp1_wt  = (unsigned short*)carve((size_t)Ck * HIDk * 2);
  unsigned short* mlp2_wt  = (unsigned short*)carve((size_t)HIDk * Ck * 2);
  (void)ws_size; (void)in_sizes; (void)n_in; (void)out_size;

  // 0) weight prep: transpose + bf16
  wt_transpose_kernel<<<(Ck * 3 * Ck + 255) / 256, 256, 0, stream>>>(qkv_w,  qkv_wt,  Ck,   3 * Ck);
  wt_transpose_kernel<<<(Ck * Ck     + 255) / 256, 256, 0, stream>>>(proj_w, proj_wt, Ck,   Ck);
  wt_transpose_kernel<<<(Ck * HIDk   + 255) / 256, 256, 0, stream>>>(mlp_w1, mlp1_wt, Ck,   HIDk);
  wt_transpose_kernel<<<(HIDk * Ck   + 255) / 256, 256, 0, stream>>>(mlp_w2, mlp2_wt, HIDk, Ck);

  // 1) dwconv residual + roll + partition
  {
    int total = NWIN * NTOK * Ck;
    dwconv_partition_kernel<<<(total + 255) / 256, 256, 0, stream>>>(x, convw, windows);
  }
  // 2) LN1 -> bf16
  layernorm_kernel<<<MROWS / 8, 256, 0, stream>>>(windows, n1g, n1b, ln1);

  // 3) QKV GEMM: [131072x192] @ [192x576], scatter q/k/v^T as bf16
  {
    int waves = (MROWS / 32) * ((3 * Ck) / 64);
    gemm_bf16_kernel<<<waves / 4, 128, 0, stream>>>(
        ln1, qkv_wt, qkv_b, MROWS, 3 * Ck, Ck, GM_QKV,
        nullptr, nullptr, nullptr, qbuf, kbuf, vtbuf);
  }
  // 4) attention per (window, head)
  attention_kernel<<<NWIN * NHEAD, 128, 0, stream>>>(qbuf, kbuf, vtbuf, relbias, attn_out);

  // 5) proj GEMM + residual -> windows2 (f32)
  {
    int waves = (MROWS / 32) * (Ck / 64);
    gemm_bf16_kernel<<<waves / 4, 128, 0, stream>>>(
        attn_out, proj_wt, proj_b, MROWS, Ck, Ck, GM_F32,
        windows2, nullptr, windows, nullptr, nullptr, nullptr);
  }
  // 6) LN2 -> bf16
  layernorm_kernel<<<MROWS / 8, 256, 0, stream>>>(windows2, n2g, n2b, ln2);

  // 7) MLP1 GEMM + GELU -> bf16
  {
    int waves = (MROWS / 32) * (HIDk / 64);
    gemm_bf16_kernel<<<waves / 4, 128, 0, stream>>>(
        ln2, mlp1_wt, mlp_b1, MROWS, HIDk, Ck, GM_GELU,
        nullptr, hbuf, nullptr, nullptr, nullptr, nullptr);
  }
  // 8) MLP2 GEMM + residual -> windows3 (f32)
  {
    int waves = (MROWS / 32) * (Ck / 64);
    gemm_bf16_kernel<<<waves / 4, 128, 0, stream>>>(
        hbuf, mlp2_wt, mlp_b2, MROWS, Ck, HIDk, GM_F32,
        windows3, nullptr, windows2, nullptr, nullptr, nullptr);
  }
  // 9) unpartition + roll back -> NCHW f32 output
  {
    int total = Bk * Ck * Hk * Wk;
    unpartition_kernel<<<(total + 255) / 256, 256, 0, stream>>>(windows3, out);
  }
}
